// ConvCrossAttention_75651553952168
// MI455X (gfx1250) — matrix operand order repaired and checked by reading, not compiled, not run on hardware
//
#include <hip/hip_runtime.h>

// ---------------------------------------------------------------------------
// Problem constants (from the reference)
// ---------------------------------------------------------------------------
#define BATCH   16
#define DIM     256
#define IMG     32            // H = W = 32
#define HEADS   8
#define DHEAD   64
#define INNER   512           // HEADS*DHEAD
#define NQ      1024          // 32*32 query pixels
#define NKV     256           // 16*16 kv pixels (stride 2)
#define EPSBN   1e-5f
#define SCALE   0.125f        // DHEAD^-0.5

typedef __attribute__((ext_vector_type(16))) _Float16 v16h;
typedef __attribute__((ext_vector_type(8)))  float    v8f;
typedef __attribute__((ext_vector_type(4)))  unsigned int u32x4;
typedef __attribute__((ext_vector_type(8)))  int      i32x8;
typedef __attribute__((ext_vector_type(4)))  int      i32x4;

union Frag16 {
    uint4 u[2];
    v16h  h;
};

__device__ __forceinline__ v8f wmma_f16(v16h a, v16h b, v8f c) {
    // D = A(16x32 f16) * B(32x16 f16) + C(16x16 f32)
    return __builtin_amdgcn_wmma_f32_16x16x32_f16(
        /*neg_a=*/false, a, /*neg_b=*/false, b,
        /*c_mod=*/(short)0, c, /*reuse_a=*/false, /*reuse_b=*/false);
}

// ---------------------------------------------------------------------------
// Tensor Data Mover: stage a 2-D tile of `rows` rows x K f16 elements
// (row stride = K elements) from global memory into LDS at byte offset
// ldsByteOff.  D# built per CDNA5 ISA (group0 + group1,
// groups 2/3 zero / unused for a 2-D tensor).  Completion via TENSORcnt.
// 6-arg builtin form (clang-23 / therock-10.0 toolchain).
// ---------------------------------------------------------------------------
__device__ __forceinline__ void tdm_load_rows_to_lds(const _Float16* gsrc,
                                                     unsigned ldsByteOff,
                                                     int K, int rows,
                                                     int totalRows) {
    unsigned long long ga = (unsigned long long)(uintptr_t)gsrc;
    u32x4 g0;
    g0[0] = 1u;                                    // count = 1 valid descriptor
    g0[1] = ldsByteOff;                            // lds_addr (bytes)
    g0[2] = (unsigned)ga;                          // global_addr[31:0]
    g0[3] = (unsigned)((ga >> 32) & 0x01FFFFFFu)   // global_addr[56:32]
          | 0x80000000u;                           // type = 2 ("image")
    i32x8 g1;
    g1[0] = (int)(1u << 16);                       // data_size = 1 -> 2 bytes
    g1[1] = (int)(((unsigned)K & 0xFFFFu) << 16);  // tensor_dim0[15:0]
    g1[2] = (int)(((unsigned)K >> 16) |
                  (((unsigned)totalRows & 0xFFFFu) << 16));   // dim0 hi | dim1 lo
    g1[3] = (int)(((unsigned)totalRows >> 16) |
                  (((unsigned)K & 0xFFFFu) << 16));           // dim1 hi | tile_dim0 = K
    g1[4] = rows;                                  // tile_dim1 = rows, tile_dim2 = 0
    g1[5] = K;                                     // tensor_dim0_stride[31:0]
    g1[6] = 0;                                     // stride hi / dim1_stride lo
    g1[7] = 0;
    i32x4 z4 = {};
    i32x8 z8 = {};
    __builtin_amdgcn_tensor_load_to_lds(g0, g1, z4, z4, z8, 0);
}

// ---------------------------------------------------------------------------
// fp32 -> f16 weight conversion
// ---------------------------------------------------------------------------
__global__ void cvt_f32_to_f16(const float* __restrict__ src,
                               _Float16* __restrict__ dst, int n) {
    int i = blockIdx.x * 256 + threadIdx.x;
    if (i < n) dst[i] = (_Float16)src[i];
}

// ---------------------------------------------------------------------------
// Depthwise 3x3 conv + folded BN -> f16 activation in [pixel, cin] layout
// ---------------------------------------------------------------------------
__global__ void dwconv_bn_f16(const float* __restrict__ x,
                              const float* __restrict__ dw,
                              const float* __restrict__ gam,
                              const float* __restrict__ bet,
                              const float* __restrict__ rm,
                              const float* __restrict__ rv,
                              _Float16* __restrict__ y,
                              int OH, int OW, int stride) {
    int tid = blockIdx.x * 256 + threadIdx.x;
    int total = BATCH * DIM * OH * OW;
    if (tid >= total) return;
    int ow = tid % OW;
    int t1 = tid / OW;
    int oh = t1 % OH;
    int t2 = t1 / OH;
    int c  = t2 % DIM;
    int b  = t2 / DIM;

    const float* xc = x + ((size_t)(b * DIM + c) * IMG) * IMG;
    const float* wc = dw + c * 9;
    float s = 0.f;
#pragma unroll
    for (int ky = 0; ky < 3; ++ky) {
        int ih = oh * stride - 1 + ky;
        if (ih < 0 || ih >= IMG) continue;
#pragma unroll
        for (int kx = 0; kx < 3; ++kx) {
            int iw = ow * stride - 1 + kx;
            if (iw < 0 || iw >= IMG) continue;
            s += xc[ih * IMG + iw] * wc[ky * 3 + kx];
        }
    }
    float inv = gam[c] * rsqrtf(rv[c] + EPSBN);
    float val = s * inv + (bet[c] - rm[c] * inv);
    y[((size_t)((b * OH + oh) * OW + ow)) * DIM + c] = (_Float16)val;
}

// ---------------------------------------------------------------------------
// Generic WMMA GEMM:  C[M,N] = A[M,K] * Bt[N,K]^T      (A,Bt f16, acc f32)
// Block = 256 threads = 8 waves; wave w owns rows m0 = bx*128 + w*16,
// cols n0 = by*64 (4 tiles of 16).
// The 64-row Bt tile is staged to LDS once per block by the Tensor Data
// Mover (wave 0 issues tensor_load_to_lds, waits TENSORcnt, then barrier);
// all 8 waves then read B fragments from LDS (ds_read_b128).
// Store layout selected by MODE:
//   MODE 0 : q tensor     half[(b*8+h)*NQ  + n][d]
//   MODE 1 : k/v tensors  half[(b*8+h)*NKV + n][d]   (c<512 -> k, else v)
//   MODE 2 : fp32 NCHW out + bias:  out[(b*DIM + c)*NQ + n]
// ---------------------------------------------------------------------------
template <int MODE>
__global__ void wmma_gemm(const _Float16* __restrict__ A,
                          const _Float16* __restrict__ Bt,
                          int K,
                          _Float16* __restrict__ outH0,
                          _Float16* __restrict__ outH1,
                          float* __restrict__ outF,
                          const float* __restrict__ bias) {
    extern __shared__ char smem_g[];          // 64 * K * 2 bytes (Bt tile)
    const int lane = threadIdx.x & 31;
    const int wv   = threadIdx.x >> 5;
    const int m0   = blockIdx.x * 128 + wv * 16;
    const int n0   = blockIdx.y * 64;
    const int mr   = lane & 15;
    const int hi   = lane >> 4;
    const int ka   = hi ? 16 : 0;   // A fragment K chunk base (contiguous layout)
    const int kb   = hi ? 16 : 0;   // B fragment K chunk base (contiguous layout)

    // --- TDM: stage Bt rows [n0, n0+64) x K into LDS (wave 0 only) ---
    if (threadIdx.x < 32) {
        tdm_load_rows_to_lds(Bt + (size_t)n0 * K, 0, K, 64, (int)gridDim.y * 64);
        __builtin_amdgcn_s_wait_tensorcnt(0);
    }
    __syncthreads();
    const _Float16* Blds = (const _Float16*)smem_g;

    v8f acc[4] = {};
    const _Float16* arow = A + (size_t)(m0 + mr) * K;

    for (int k = 0; k < K; k += 32) {
        if ((k & 127) == 0)
            __builtin_prefetch(arow + k + 128, 0, 1);   // global_prefetch_b8
        Frag16 af;
        af.u[0] = *(const uint4*)(arow + k + ka);
        af.u[1] = *(const uint4*)(arow + k + ka + 8);
#pragma unroll
        for (int t = 0; t < 4; ++t) {
            const _Float16* brow = Blds + (size_t)(t * 16 + mr) * K;
            Frag16 bf;
            bf.u[0] = *(const uint4*)(brow + k + kb);
            bf.u[1] = *(const uint4*)(brow + k + kb + 8);
            acc[t] = wmma_f16(af.h, bf.h, acc[t]);
        }
    }

#pragma unroll
    for (int t = 0; t < 4; ++t) {
        int c = n0 + t * 16 + mr;
#pragma unroll
        for (int r = 0; r < 8; ++r) {
            int m = m0 + r + 8 * hi;
            float v = acc[t][r];
            if (MODE == 0) {
                int b = m >> 10, np = m & 1023;
                int h = c >> 6,  d  = c & 63;
                outH0[(((size_t)(b * HEADS + h)) * NQ + np) * DHEAD + d] = (_Float16)v;
            } else if (MODE == 1) {
                int b = m >> 8, np = m & 255;
                int cc = (c < INNER) ? c : c - INNER;
                _Float16* dst = (c < INNER) ? outH0 : outH1;
                dst[(((size_t)(b * HEADS + (cc >> 6))) * NKV + np) * DHEAD + (cc & 63)] =
                    (_Float16)v;
            } else {
                int b = m >> 10, np = m & 1023;
                outF[((size_t)(b * DIM + c)) * NQ + np] = v + bias[c];
            }
        }
    }
}

// ---------------------------------------------------------------------------
// Cross attention for one (b, h) and a 64-query tile.
// block = 128 threads = 4 waves (wave w -> query rows w*16..w*16+15)
// dynamic LDS: scores f32 [64][256] | P f16 [64][256] | Vt f16 [64][256]
// ---------------------------------------------------------------------------
__global__ void attention_kernel(const _Float16* __restrict__ q,
                                 const _Float16* __restrict__ kt,
                                 const _Float16* __restrict__ vt,
                                 _Float16* __restrict__ omerged) {
    extern __shared__ char smem[];
    float*    s_sc = (float*)smem;                                  // 64*256*4
    _Float16* s_p  = (_Float16*)(smem + 64 * 256 * 4);              // 64*256*2
    _Float16* s_vt = (_Float16*)(smem + 64 * 256 * 4 + 64 * 256 * 2);

    const int bh    = blockIdx.y;
    const int b     = bh >> 3;
    const int h     = bh & 7;
    const int qBase = blockIdx.x * 64;
    const int lane  = threadIdx.x & 31;
    const int wv    = threadIdx.x >> 5;
    const int mr    = lane & 15;
    const int hi    = lane >> 4;
    const int ka    = hi ? 16 : 0;
    const int kb    = hi ? 16 : 0;

    const size_t headQ  = ((size_t)(b * HEADS + h)) * NQ  * DHEAD;
    const size_t headKV = ((size_t)(b * HEADS + h)) * NKV * DHEAD;

    // --- stage V transposed into LDS:  s_vt[d][n] ---
    for (int i = threadIdx.x; i < NKV * DHEAD; i += 128) {
        int n = i >> 6, d = i & 63;
        s_vt[d * NKV + n] = vt[headKV + i];
    }

    // --- phase 1: S = Q K^T * scale  (K=64 -> two 16x16x32 WMMAs per tile) ---
    const _Float16* qrow = q + headQ + (size_t)(qBase + wv * 16 + mr) * DHEAD;
    Frag16 a0, a1;
    a0.u[0] = *(const uint4*)(qrow + ka);
    a0.u[1] = *(const uint4*)(qrow + ka + 8);
    a1.u[0] = *(const uint4*)(qrow + 32 + ka);
    a1.u[1] = *(const uint4*)(qrow + 32 + ka + 8);

    for (int nt = 0; nt < 16; ++nt) {
        const _Float16* krow = kt + headKV + (size_t)(nt * 16 + mr) * DHEAD;
        Frag16 b0, b1;
        b0.u[0] = *(const uint4*)(krow + kb);
        b0.u[1] = *(const uint4*)(krow + kb + 8);
        b1.u[0] = *(const uint4*)(krow + 32 + kb);
        b1.u[1] = *(const uint4*)(krow + 32 + kb + 8);
        v8f acc = {};
        acc = wmma_f16(a0.h, b0.h, acc);
        acc = wmma_f16(a1.h, b1.h, acc);
        int col   = nt * 16 + mr;
        int rbase = wv * 16 + 8 * hi;
#pragma unroll
        for (int r = 0; r < 8; ++r)
            s_sc[(rbase + r) * NKV + col] = acc[r] * SCALE;
    }
    __syncthreads();

    // --- softmax over 256 keys, one thread per query row ---
    if (threadIdx.x < 64) {
        float* row = s_sc + threadIdx.x * NKV;
        float mx = -3.4e38f;
        for (int j = 0; j < NKV; ++j) mx = fmaxf(mx, row[j]);
        float sum = 0.f;
        for (int j = 0; j < NKV; ++j) {
            float e = __expf(row[j] - mx);
            row[j] = e;
            sum += e;
        }
        float rinv = 1.f / sum;
        _Float16* prow = s_p + threadIdx.x * NKV;
        for (int j = 0; j < NKV; ++j) prow[j] = (_Float16)(row[j] * rinv);
    }
    __syncthreads();

    // --- phase 2: O = P V   (P:16x256 f16 from LDS, Vt staged in LDS) ---
    v8f oacc[4] = {};
    const _Float16* prow = s_p + (size_t)(wv * 16 + mr) * NKV;
    for (int k = 0; k < NKV; k += 32) {
        Frag16 pf;
        pf.u[0] = *(const uint4*)(prow + k + ka);
        pf.u[1] = *(const uint4*)(prow + k + ka + 8);
#pragma unroll
        for (int t = 0; t < 4; ++t) {
            const _Float16* vrow = s_vt + (size_t)(t * 16 + mr) * NKV;
            Frag16 vf;
            vf.u[0] = *(const uint4*)(vrow + k + kb);
            vf.u[1] = *(const uint4*)(vrow + k + kb + 8);
            oacc[t] = wmma_f16(pf.h, vf.h, oacc[t]);
        }
    }

#pragma unroll
    for (int t = 0; t < 4; ++t) {
        int d = t * 16 + mr;
#pragma unroll
        for (int r = 0; r < 8; ++r) {
            int n = qBase + wv * 16 + r + 8 * hi;
            omerged[((size_t)(b * NQ + n)) * INNER + h * DHEAD + d] =
                (_Float16)oacc[t][r];
        }
    }
}

// ---------------------------------------------------------------------------
// host-side launch
// ---------------------------------------------------------------------------
extern "C" void kernel_launch(void* const* d_in, const int* in_sizes, int n_in,
                              void* d_out, int out_size, void* d_ws, size_t ws_size,
                              hipStream_t stream) {
    (void)in_sizes; (void)n_in; (void)out_size; (void)ws_size;

    const float* x1 = (const float*)d_in[0];
    const float* x2 = (const float*)d_in[1];
    // per-branch params: dw, g, b, rm, rv, pw
    const float* q1p[6]  = {(const float*)d_in[2],  (const float*)d_in[3],
                            (const float*)d_in[4],  (const float*)d_in[5],
                            (const float*)d_in[6],  (const float*)d_in[7]};
    const float* kv1p[6] = {(const float*)d_in[8],  (const float*)d_in[9],
                            (const float*)d_in[10], (const float*)d_in[11],
                            (const float*)d_in[12], (const float*)d_in[13]};
    const float* q2p[6]  = {(const float*)d_in[14], (const float*)d_in[15],
                            (const float*)d_in[16], (const float*)d_in[17],
                            (const float*)d_in[18], (const float*)d_in[19]};
    const float* kv2p[6] = {(const float*)d_in[20], (const float*)d_in[21],
                            (const float*)d_in[22], (const float*)d_in[23],
                            (const float*)d_in[24], (const float*)d_in[25]};
    const float* out1_w = (const float*)d_in[26];
    const float* out1_b = (const float*)d_in[27];
    const float* out2_w = (const float*)d_in[28];
    const float* out2_b = (const float*)d_in[29];

    // ---- workspace carving (256B aligned) ----
    char* ws = (char*)d_ws;
    size_t off = 0;
    auto carve = [&](size_t bytes) -> char* {
        char* p = ws + off;
        off += (bytes + 255) & ~(size_t)255;
        return p;
    };
    _Float16* wq1h  = (_Float16*)carve((size_t)INNER * DIM * 2);       // 512x256
    _Float16* wkv1h = (_Float16*)carve((size_t)2 * INNER * DIM * 2);   // 1024x256
    _Float16* wq2h  = (_Float16*)carve((size_t)INNER * DIM * 2);
    _Float16* wkv2h = (_Float16*)carve((size_t)2 * INNER * DIM * 2);
    _Float16* wo1h  = (_Float16*)carve((size_t)DIM * INNER * 2);       // 256x512
    _Float16* wo2h  = (_Float16*)carve((size_t)DIM * INNER * 2);
    _Float16* yq1   = (_Float16*)carve((size_t)BATCH * NQ  * DIM * 2);
    _Float16* yq2   = (_Float16*)carve((size_t)BATCH * NQ  * DIM * 2);
    _Float16* ykv1  = (_Float16*)carve((size_t)BATCH * NKV * DIM * 2);
    _Float16* ykv2  = (_Float16*)carve((size_t)BATCH * NKV * DIM * 2);
    _Float16* q1    = (_Float16*)carve((size_t)BATCH * INNER * NQ  * 2);
    _Float16* q2    = (_Float16*)carve((size_t)BATCH * INNER * NQ  * 2);
    _Float16* k1    = (_Float16*)carve((size_t)BATCH * INNER * NKV * 2);
    _Float16* v1    = (_Float16*)carve((size_t)BATCH * INNER * NKV * 2);
    _Float16* k2    = (_Float16*)carve((size_t)BATCH * INNER * NKV * 2);
    _Float16* v2    = (_Float16*)carve((size_t)BATCH * INNER * NKV * 2);
    _Float16* o1m   = (_Float16*)carve((size_t)BATCH * NQ * INNER * 2);
    _Float16* o2m   = (_Float16*)carve((size_t)BATCH * NQ * INNER * 2);

    // ---- weight conversions ----
    auto cvt = [&](const float* src, _Float16* dst, int n) {
        cvt_f32_to_f16<<<(n + 255) / 256, 256, 0, stream>>>(src, dst, n);
    };
    cvt(q1p[5],  wq1h,  INNER * DIM);
    cvt(kv1p[5], wkv1h, 2 * INNER * DIM);
    cvt(q2p[5],  wq2h,  INNER * DIM);
    cvt(kv2p[5], wkv2h, 2 * INNER * DIM);
    cvt(out1_w,  wo1h,  DIM * INNER);
    cvt(out2_w,  wo2h,  DIM * INNER);

    // ---- depthwise + BN -> f16 ----
    {
        int totQ = BATCH * DIM * IMG * IMG;          // stride 1, 32x32
        int totK = BATCH * DIM * 16 * 16;            // stride 2, 16x16
        dwconv_bn_f16<<<(totQ + 255) / 256, 256, 0, stream>>>(
            x1, q1p[0], q1p[1], q1p[2], q1p[3], q1p[4], yq1, IMG, IMG, 1);
        dwconv_bn_f16<<<(totK + 255) / 256, 256, 0, stream>>>(
            x1, kv1p[0], kv1p[1], kv1p[2], kv1p[3], kv1p[4], ykv1, 16, 16, 2);
        dwconv_bn_f16<<<(totQ + 255) / 256, 256, 0, stream>>>(
            x2, q2p[0], q2p[1], q2p[2], q2p[3], q2p[4], yq2, IMG, IMG, 1);
        dwconv_bn_f16<<<(totK + 255) / 256, 256, 0, stream>>>(
            x2, kv2p[0], kv2p[1], kv2p[2], kv2p[3], kv2p[4], ykv2, 16, 16, 2);
    }

    // ---- pointwise projections (WMMA GEMMs, B tile staged by TDM) ----
    size_t shB256 = (size_t)64 * 256 * 2;   // 32 KB
    size_t shB512 = (size_t)64 * 512 * 2;   // 64 KB
    // q: M = B*NQ = 16384, N = 512, K = 256
    wmma_gemm<0><<<dim3(16384 / 128, INNER / 64), 256, shB256, stream>>>(
        yq1, wq1h, DIM, q1, nullptr, nullptr, nullptr);
    wmma_gemm<0><<<dim3(16384 / 128, INNER / 64), 256, shB256, stream>>>(
        yq2, wq2h, DIM, q2, nullptr, nullptr, nullptr);
    // kv: M = B*NKV = 4096, N = 1024, K = 256
    wmma_gemm<1><<<dim3(4096 / 128, (2 * INNER) / 64), 256, shB256, stream>>>(
        ykv1, wkv1h, DIM, k1, v1, nullptr, nullptr);
    wmma_gemm<1><<<dim3(4096 / 128, (2 * INNER) / 64), 256, shB256, stream>>>(
        ykv2, wkv2h, DIM, k2, v2, nullptr, nullptr);

    // ---- cross attention (q1 attends k2/v2; q2 attends k1/v1) ----
    {
        size_t shmem = (size_t)64 * 256 * 4 + 2 * (size_t)64 * 256 * 2; // 128 KB
        dim3 grid(NQ / 64, BATCH * HEADS);
        attention_kernel<<<grid, 128, shmem, stream>>>(q1, k2, v2, o1m);
        attention_kernel<<<grid, 128, shmem, stream>>>(q2, k1, v1, o2m);
    }

    // ---- output projections: M = 16384, N = 256, K = 512, fp32 out + bias ----
    float* out = (float*)d_out;
    wmma_gemm<2><<<dim3(16384 / 128, DIM / 64), 256, shB512, stream>>>(
        o1m, wo1h, INNER, nullptr, nullptr, out, out1_b);
    wmma_gemm<2><<<dim3(16384 / 128, DIM / 64), 256, shB512, stream>>>(
        o2m, wo2h, INNER, nullptr, nullptr,
        out + (size_t)BATCH * DIM * NQ, out2_b);
}
